// LinearAttentionEncoder_52407190946369
// MI455X (gfx1250) — matrix operand order repaired and checked by reading, not compiled, not run on hardware
//
#include <hip/hip_runtime.h>

// Model dims (fixed by the reference)
#define DM 256
#define NH 8
#define HD 32
#define NL 4
#define FF 1024
#define CSTRIDE 1000
#define NB 4
#define MMEL 128
#define TT 16000
#define BT (NB * TT)          // 64000 rows
#define KSPLIT 32             // split-K ways for the conv GEMM

typedef _Float16 half8 __attribute__((ext_vector_type(8)));
typedef _Float16 v16h  __attribute__((ext_vector_type(16)));
typedef float    v8f   __attribute__((ext_vector_type(8)));

// LDS strides in halves: >=32 and 16B-aligned row bases (40*2=80 bytes)
#define AS_STRIDE 40
#define BS_STRIDE 40

// ---------------------------------------------------------------------------
// Activation epilogues: 0=none, 1=elu(x)+1, 2=exact gelu
// ---------------------------------------------------------------------------
template <int ACT>
__device__ __forceinline__ float act_apply(float x) {
  if (ACT == 1) return x > 0.f ? x + 1.f : expf(x);            // elu+1
  if (ACT == 2) return 0.5f * x * (1.f + erff(x * 0.7071067811865475f));
  return x;
}

// ---------------------------------------------------------------------------
// Generic tiled WMMA GEMM:  C[M x N] = act(A[M x K] * W + bias) (+ Cres)
//   A: f32 or f16, row stride lda.  WT: f16 weights PRE-TRANSPOSED to [N][K]
//   so B-tile staging is a contiguous b128 copy (no scalar DS scatter).
//   Block = 256 threads (8 waves). Block tile 64(M) x 128(N); wave tile 16x64
//   (four 16x16x32 WMMA accumulators per wave -> one A fragment feeds 4 WMMAs).
//   Software pipelined: next K-tile's global loads are issued into registers
//   while the current LDS tile is consumed by the matrix pipe.
//   SPLIT: grid.z-way split-K, each z writes a partial [64 x N] tile
//   (bias/act/res skipped; folded by conv_reduce_kernel afterwards).
//   Requirements: M%64==0, N%128==0, K%(32*grid.z)==0 (all call sites comply).
// ---------------------------------------------------------------------------
template <typename AT, int ACT, bool RES, bool SPLIT>
__global__ __launch_bounds__(256) void gemm_wmma_kernel(
    const AT* __restrict__ A, long lda,
    const _Float16* __restrict__ WT,
    const float* __restrict__ bias,
    const float* __restrict__ Cres,
    float* __restrict__ C,
    int N, long Kdim)
{
  __shared__ _Float16 As[64 * AS_STRIDE];     // A tile 64(M) x 32(K)
  __shared__ _Float16 BsT[128 * BS_STRIDE];   // B tile: [n][k]

  const int t    = threadIdx.x;
  const int lane = t & 31;
  const int wid  = t >> 5;
  const int wm   = wid & 3;   // which 16-row stripe (4)
  const int wn   = wid >> 2;  // which 64-col stripe (2)
  const long rowBase = (long)blockIdx.x * 64;
  const int  colBase = blockIdx.y * 128;

  const long kPer = Kdim / (long)gridDim.z;
  const long kbeg = kPer * (long)blockIdx.z;

  v8f acc0 = {}, acc1 = {}, acc2 = {}, acc3 = {};

  // A-stage map: thread -> (row 0..63, col {0,8,16,24}), 8 contiguous K each
  const int ar = t >> 2;
  const int ac = (t & 3) * 8;
  // B-stage map: [N][K] source -> thread handles n-row (t>>1), 16 contiguous k
  const int bn  = t >> 1;         // n row 0..127
  const int bkc = (t & 1) * 16;   // k col base {0,16}

  const int lm = lane & 15;
  const int hi = lane >> 4;           // half-wave selector
  const int a_kb_lo = hi ? 16 : 0;    // A-layout (matches B): lanes16-31 hold K=16..23
  const int a_kb_hi = hi ? 24 : 8;    //               and K=24..31
  const int b_kb    = hi ? 16 : 0;    // contiguous-K B-layout (matches A): lanes16-31 hold K=16..31

  // Pipelined staging registers
  AT       aReg[8];
  _Float16 bReg[16];
  {
    const AT* srcA = A + (rowBase + ar) * lda + kbeg + ac;
#pragma unroll
    for (int i = 0; i < 8; ++i) aReg[i] = srcA[i];
    const _Float16* srcB = WT + (long)(colBase + bn) * Kdim + kbeg + bkc;
#pragma unroll
    for (int i = 0; i < 16; ++i) bReg[i] = srcB[i];
  }

  for (long k0 = kbeg; k0 < kbeg + kPer; k0 += 32) {
    // Commit staged registers to LDS (convert A to f16; B is a straight copy)
    {
      _Float16* dstA = &As[ar * AS_STRIDE + ac];
#pragma unroll
      for (int i = 0; i < 8; ++i) dstA[i] = (_Float16)aReg[i];
      _Float16* dstB = &BsT[bn * BS_STRIDE + bkc];
#pragma unroll
      for (int i = 0; i < 16; ++i) dstB[i] = bReg[i];
    }
    __syncthreads();

    // Issue next tile's global loads; they complete while WMMAs run below.
    const long kn = k0 + 32;
    if (kn < kbeg + kPer) {
      const AT* srcA = A + (rowBase + ar) * lda + kn + ac;
      const _Float16* srcB = WT + (long)(colBase + bn) * Kdim + kn + bkc;
      // Prefetch one more tile ahead (global_prefetch_b8) to prime L2 streams.
      if (kn + 32 < kbeg + kPer) {
        __builtin_prefetch(srcA + 32, 0, 1);
        __builtin_prefetch(srcB + 32, 0, 1);
      }
#pragma unroll
      for (int i = 0; i < 8; ++i) aReg[i] = srcA[i];
#pragma unroll
      for (int i = 0; i < 16; ++i) bReg[i] = srcB[i];
    }

    // Build A fragment (16x32 f16): contiguous-K vector loads from LDS
    v16h afrag;
    {
      const _Float16* ap = &As[(wm * 16 + lm) * AS_STRIDE];
      half8 alo = *(const half8*)(ap + a_kb_lo);
      half8 ahi = *(const half8*)(ap + a_kb_hi);
#pragma unroll
      for (int i = 0; i < 8; ++i) { afrag[i] = alo[i]; afrag[i + 8] = ahi[i]; }
    }
    // Build 4 B fragments (32x16 f16 each): contiguous in BsT
    v16h bfrag[4];
#pragma unroll
    for (int j = 0; j < 4; ++j) {
      const _Float16* bp = &BsT[(wn * 64 + j * 16 + lm) * BS_STRIDE + b_kb];
      half8 ba = *(const half8*)(bp);
      half8 bb = *(const half8*)(bp + 8);
#pragma unroll
      for (int i = 0; i < 8; ++i) { bfrag[j][i] = ba[i]; bfrag[j][i + 8] = bb[i]; }
    }

    acc0 = __builtin_amdgcn_wmma_f32_16x16x32_f16(false, afrag, false, bfrag[0],
                                                  (short)0, acc0, false, false);
    acc1 = __builtin_amdgcn_wmma_f32_16x16x32_f16(false, afrag, false, bfrag[1],
                                                  (short)0, acc1, false, false);
    acc2 = __builtin_amdgcn_wmma_f32_16x16x32_f16(false, afrag, false, bfrag[2],
                                                  (short)0, acc2, false, false);
    acc3 = __builtin_amdgcn_wmma_f32_16x16x32_f16(false, afrag, false, bfrag[3],
                                                  (short)0, acc3, false, false);
    __syncthreads();
  }

  // Epilogue per ISA C/D layout: VGPR r -> M = (lane<16 ? r : r+8), N = lane&15
  v8f accs[4] = {acc0, acc1, acc2, acc3};
#pragma unroll
  for (int j = 0; j < 4; ++j) {
    const int col = colBase + wn * 64 + j * 16 + lm;
    const float bcol = SPLIT ? 0.f : bias[col];
#pragma unroll
    for (int r = 0; r < 8; ++r) {
      long row = rowBase + wm * 16 + (hi ? r + 8 : r);
      if (SPLIT) {
        C[((long)blockIdx.z * 64 + row) * N + col] = accs[j][r];
      } else {
        float v0 = act_apply<ACT>(accs[j][r] + bcol);
        if (RES) v0 += Cres[row * (long)N + col];
        C[row * (long)N + col] = v0;
      }
    }
  }
}

// ---------------------------------------------------------------------------
// Split-K reduction for the conv GEMM: g0 = sum_z partial[z] + conv_b
// ---------------------------------------------------------------------------
__global__ __launch_bounds__(256) void conv_reduce_kernel(
    const float* __restrict__ part, const float* __restrict__ bias,
    float* __restrict__ g0)
{
  int idx = blockIdx.x * 256 + threadIdx.x;   // over 64*256
  int o   = idx & (DM - 1);
  float s = bias[o];
  for (int z = 0; z < KSPLIT; ++z) s += part[(long)z * 64 * DM + idx];
  g0[idx] = s;
}

// ---------------------------------------------------------------------------
// LayerNorm over D=256 (one row per block, biased variance like jnp.var)
// ---------------------------------------------------------------------------
__global__ __launch_bounds__(256) void layernorm_kernel(
    const float* __restrict__ x, const float* __restrict__ w,
    const float* __restrict__ b, float* __restrict__ out)
{
  __shared__ float r1[256];
  __shared__ float r2[256];
  long row = blockIdx.x;
  int  t   = threadIdx.x;
  float v  = x[row * DM + t];
  r1[t] = v; r2[t] = v * v;
  __syncthreads();
  for (int s = 128; s > 0; s >>= 1) {
    if (t < s) { r1[t] += r1[t + s]; r2[t] += r2[t + s]; }
    __syncthreads();
  }
  float mean = r1[0] * (1.f / 256.f);
  float var  = r2[0] * (1.f / 256.f) - mean * mean;
  float inv  = rsqrtf(var + 1e-5f);
  out[row * DM + t] = (v - mean) * inv * w[t] + b[t];
}

// ---------------------------------------------------------------------------
// Sinusoidal positional encoding add (interleaved sin/cos)
// ---------------------------------------------------------------------------
__global__ void posadd_kernel(float* __restrict__ x,
                              const float* __restrict__ pos_scale)
{
  long idx = (long)blockIdx.x * 256 + threadIdx.x;
  int  d   = (int)(idx & (DM - 1));
  long bt  = idx >> 8;
  int  tp  = (int)(bt % TT);
  int  i2  = d >> 1;
  float freq = expf((float)(2 * i2) * (-9.210340371976184f / (float)DM)); // -ln(1e4)/D
  float ang  = (float)tp * freq;
  float pe   = (d & 1) ? cosf(ang) : sinf(ang);
  x[idx] += pe * pos_scale[0];
}

// ---------------------------------------------------------------------------
// Linear attention state: kv[b,h] = sum_n k^T v (32x32), ksum[b,h] = sum_n k
// One block per (b,h); k/v are (B,T,D) with head h in cols h*32..h*32+31.
// ---------------------------------------------------------------------------
__global__ __launch_bounds__(256) void attn_kv_kernel(
    const float* __restrict__ k, const float* __restrict__ v,
    float* __restrict__ kvout, float* __restrict__ ksumout)
{
  int bh = blockIdx.x;            // 0..31
  int b  = bh >> 3, h = bh & 7;
  int t  = threadIdx.x;
  int d  = t >> 3;                // 0..31
  int eb = (t & 7) * 4;           // e base
  __shared__ float kb[8][32];
  __shared__ float vb[8][32];
  float acc[4] = {0.f, 0.f, 0.f, 0.f};
  float ks = 0.f;
  const float* kbase = k + (long)b * TT * DM + h * HD;
  const float* vbase = v + (long)b * TT * DM + h * HD;
  int sr = t >> 5, sc = t & 31;
  for (int n0 = 0; n0 < TT; n0 += 8) {
    kb[sr][sc] = kbase[(long)(n0 + sr) * DM + sc];
    vb[sr][sc] = vbase[(long)(n0 + sr) * DM + sc];
    __syncthreads();
#pragma unroll
    for (int r = 0; r < 8; ++r) {
      float kd = kb[r][d];
#pragma unroll
      for (int j = 0; j < 4; ++j) acc[j] += kd * vb[r][eb + j];
      ks += kd;
    }
    __syncthreads();
  }
#pragma unroll
  for (int j = 0; j < 4; ++j)
    kvout[((long)bh * HD + d) * HD + eb + j] = acc[j];
  if ((t & 7) == 0) ksumout[bh * HD + d] = ks;
}

// ---------------------------------------------------------------------------
// out[b,n,h,e] = (q . kv[h]) / (q . ksum[h] + 1e-6). 32 rows per block.
// ---------------------------------------------------------------------------
__global__ __launch_bounds__(256) void attn_out_kernel(
    const float* __restrict__ q, const float* __restrict__ kv,
    const float* __restrict__ ksum, float* __restrict__ out)
{
  int bidx  = blockIdx.x;          // B * (T/32)
  int b     = bidx / (TT / 32);
  int chunk = bidx % (TT / 32);
  int t     = threadIdx.x;
  __shared__ float kvs[NH * HD * HD];   // 32 KB
  __shared__ float kss[NH * HD];
  __shared__ float qrow[DM];
  const float* kvsrc = kv + (long)b * NH * HD * HD;
  for (int i = t; i < NH * HD * HD; i += 256) kvs[i] = kvsrc[i];
  if (t < NH * HD) kss[t] = ksum[b * NH * HD + t];
  __syncthreads();
  int h = t >> 5, e = t & 31;
  const float* kvh = &kvs[h * HD * HD];
  const float* ksh = &kss[h * HD];
  for (int r = 0; r < 32; ++r) {
    long n = (long)b * TT + chunk * 32 + r;
    qrow[t] = q[n * DM + t];
    __syncthreads();
    float dot = 0.f, nrm = 0.f;
    const float* qh = &qrow[h * HD];
#pragma unroll
    for (int d2 = 0; d2 < HD; ++d2) {
      float qd = qh[d2];
      dot += qd * kvh[d2 * HD + e];
      nrm += qd * ksh[d2];
    }
    out[n * DM + t] = dot / (nrm + 1e-6f);
    __syncthreads();
  }
}

// ---------------------------------------------------------------------------
// Prep kernels
// ---------------------------------------------------------------------------
// Weights (L layers of [K][N] f32) -> [l][n][k] f16 (transposed per layer).
__global__ void w_transpose_kernel(const float* __restrict__ w,
                                   _Float16* __restrict__ out, int K, int N)
{
  long idx = (long)blockIdx.x * 256 + threadIdx.x;  // over L*N*K, out-order
  long KN  = (long)K * N;
  int  k   = (int)(idx % K);
  long r   = idx / K;
  int  n   = (int)(r % N);
  long l   = r / N;
  out[idx] = (_Float16)w[l * KN + (long)k * N + n];
}

// mel (B,M,T) f32 -> melT (B,T,M) f16  (A-side layout)
__global__ void mel_transpose_kernel(const float* __restrict__ mel,
                                     _Float16* __restrict__ out)
{
  long i = (long)blockIdx.x * 256 + threadIdx.x;   // over B*T*M
  int  m = (int)(i & (MMEL - 1));
  long bt = i >> 7;
  int  tp = (int)(bt % TT);
  int  b  = (int)(bt / TT);
  out[i] = (_Float16)mel[((long)(b * MMEL + m)) * TT + tp];
}

// conv_w (O,I,K) f32 -> [o][k*256+i] f16  ([N][K] layout, K = 256000)
__global__ void convw_transpose_kernel(const float* __restrict__ w,
                                       _Float16* __restrict__ out)
{
  long idx = (long)blockIdx.x * 256 + threadIdx.x;  // over 256 * 256000
  long rem = idx % (long)(DM * CSTRIDE);
  int  o   = (int)(idx / (long)(DM * CSTRIDE));
  int  k   = (int)(rem >> 8);
  int  i   = (int)(rem & (DM - 1));
  out[idx] = (_Float16)w[((long)o * DM + i) * CSTRIDE + k];
}

// ---------------------------------------------------------------------------
// Host orchestration
// ---------------------------------------------------------------------------
#define GEMM(AT, ACT, RES, SPLIT, Aptr, lda, Wp, Bp, Cresp, Cp, Mv, Nv, Kv, Zv) \
  gemm_wmma_kernel<AT, ACT, RES, SPLIT>                                         \
      <<<dim3((Mv) / 64, (Nv) / 128, (Zv)), 256, 0, stream>>>(                  \
          Aptr, (long)(lda), Wp, Bp, Cresp, Cp, (Nv), (long)(Kv))

extern "C" void kernel_launch(void* const* d_in, const int* in_sizes, int n_in,
                              void* d_out, int out_size, void* d_ws,
                              size_t ws_size, hipStream_t stream)
{
  (void)in_sizes; (void)n_in; (void)out_size; (void)ws_size;
  const float* mel     = (const float*)d_in[0];
  const float* mel_w   = (const float*)d_in[1];
  const float* mel_b   = (const float*)d_in[2];
  const float* pscale  = (const float*)d_in[3];
  const float* ln1w    = (const float*)d_in[4];
  const float* ln1b    = (const float*)d_in[5];
  const float* ln2w    = (const float*)d_in[6];
  const float* ln2b    = (const float*)d_in[7];
  const float* wq      = (const float*)d_in[8];
  const float* bq      = (const float*)d_in[9];
  const float* wk      = (const float*)d_in[10];
  const float* bk      = (const float*)d_in[11];
  const float* wv      = (const float*)d_in[12];
  const float* bv      = (const float*)d_in[13];
  const float* wo      = (const float*)d_in[14];
  const float* bo      = (const float*)d_in[15];
  const float* w1      = (const float*)d_in[16];
  const float* b1      = (const float*)d_in[17];
  const float* w2      = (const float*)d_in[18];
  const float* b2      = (const float*)d_in[19];
  const float* fin_w   = (const float*)d_in[20];
  const float* fin_b   = (const float*)d_in[21];
  const float* conv_w  = (const float*)d_in[22];
  const float* conv_b  = (const float*)d_in[23];
  const float* gp_w1   = (const float*)d_in[24];
  const float* gp_b1   = (const float*)d_in[25];
  const float* gp_w2   = (const float*)d_in[26];
  const float* gp_b2   = (const float*)d_in[27];

  float* frame = (float*)d_out;                 // (B,T,D)
  float* gout  = frame + (long)BT * DM;         // (B,16,D)

  // Workspace bump allocator (256B aligned)
  char* p = (char*)d_ws;
  auto alloc = [&](size_t bytes) -> char* {
    char* r = p;
    p += (bytes + 255) & ~(size_t)255;
    return r;
  };
  float* x    = (float*)alloc((size_t)BT * DM * 4);
  float* t0   = (float*)alloc((size_t)BT * DM * 4);
  float* qb   = (float*)alloc((size_t)BT * DM * 4);
  float* kbuf = (float*)alloc((size_t)BT * DM * 4);
  float* vbuf = (float*)alloc((size_t)BT * DM * 4);
  float* h1   = (float*)alloc((size_t)BT * FF * 4);
  float* kvb  = (float*)alloc((size_t)NB * NH * HD * HD * 4);
  float* ksb  = (float*)alloc((size_t)NB * NH * HD * 4);
  float* g0   = (float*)alloc((size_t)64 * DM * 4);
  float* g1   = (float*)alloc((size_t)64 * DM * 4);
  float* g0p  = (float*)alloc((size_t)KSPLIT * 64 * DM * 4);   // split-K partials
  _Float16* wq16   = (_Float16*)alloc((size_t)NL * DM * DM * 2);
  _Float16* wk16   = (_Float16*)alloc((size_t)NL * DM * DM * 2);
  _Float16* wv16   = (_Float16*)alloc((size_t)NL * DM * DM * 2);
  _Float16* wo16   = (_Float16*)alloc((size_t)NL * DM * DM * 2);
  _Float16* w116   = (_Float16*)alloc((size_t)NL * DM * FF * 2);
  _Float16* w216   = (_Float16*)alloc((size_t)NL * FF * DM * 2);
  _Float16* melw16 = (_Float16*)alloc((size_t)MMEL * DM * 2);
  _Float16* gpw116 = (_Float16*)alloc((size_t)DM * DM * 2);
  _Float16* gpw216 = (_Float16*)alloc((size_t)DM * DM * 2);
  _Float16* cw16   = (_Float16*)alloc((size_t)DM * CSTRIDE * DM * 2);
  _Float16* melT   = (_Float16*)alloc((size_t)BT * MMEL * 2);

  // --- weight prep: f32 -> f16 with [N][K] transpose ---
  auto wprep = [&](const float* src, _Float16* dst, int K, int N, int layers) {
    long n = (long)layers * K * N;
    w_transpose_kernel<<<(unsigned)(n / 256), 256, 0, stream>>>(src, dst, K, N);
  };
  wprep(wq, wq16, DM, DM, NL);
  wprep(wk, wk16, DM, DM, NL);
  wprep(wv, wv16, DM, DM, NL);
  wprep(wo, wo16, DM, DM, NL);
  wprep(w1, w116, DM, FF, NL);
  wprep(w2, w216, FF, DM, NL);
  wprep(mel_w, melw16, MMEL, DM, 1);
  wprep(gp_w1, gpw116, DM, DM, 1);
  wprep(gp_w2, gpw216, DM, DM, 1);
  convw_transpose_kernel<<<(unsigned)(((long)DM * CSTRIDE * DM) / 256), 256, 0,
                           stream>>>(conv_w, cw16);
  mel_transpose_kernel<<<(unsigned)(((long)BT * MMEL) / 256), 256, 0, stream>>>(
      mel, melT);

  // --- mel projection + positional encoding ---
  GEMM(_Float16, 0, false, false, melT, MMEL, melw16, mel_b, nullptr, x,
       BT, DM, MMEL, 1);
  posadd_kernel<<<(unsigned)(((long)BT * DM) / 256), 256, 0, stream>>>(x, pscale);

  // --- transformer layers ---
  for (int l = 0; l < NL; ++l) {
    const long woff  = (long)l * DM * DM;
    const long w1off = (long)l * DM * FF;
    const long w2off = (long)l * FF * DM;
    layernorm_kernel<<<BT, 256, 0, stream>>>(x, ln1w + l * DM, ln1b + l * DM, t0);
    GEMM(float, 1, false, false, t0, DM, wq16 + woff, bq + l * DM, nullptr, qb,
         BT, DM, DM, 1);
    GEMM(float, 1, false, false, t0, DM, wk16 + woff, bk + l * DM, nullptr, kbuf,
         BT, DM, DM, 1);
    GEMM(float, 0, false, false, t0, DM, wv16 + woff, bv + l * DM, nullptr, vbuf,
         BT, DM, DM, 1);
    attn_kv_kernel<<<NB * NH, 256, 0, stream>>>(kbuf, vbuf, kvb, ksb);
    attn_out_kernel<<<NB * (TT / 32), 256, 0, stream>>>(qb, kvb, ksb, t0);
    GEMM(float, 0, true, false, t0, DM, wo16 + woff, bo + l * DM, x, x,
         BT, DM, DM, 1);
    layernorm_kernel<<<BT, 256, 0, stream>>>(x, ln2w + l * DM, ln2b + l * DM, t0);
    GEMM(float, 2, false, false, t0, DM, w116 + w1off, b1 + l * FF, nullptr, h1,
         BT, FF, DM, 1);
    GEMM(float, 0, true, false, h1, FF, w216 + w2off, b2 + l * DM, x, x,
         BT, DM, FF, 1);
  }

  // --- final layernorm -> frame output ---
  layernorm_kernel<<<BT, 256, 0, stream>>>(x, fin_w, fin_b, frame);

  // --- strided conv as split-K GEMM over contiguous frame rows ---
  GEMM(float, 0, false, true, frame, (long)DM * CSTRIDE, cw16, nullptr, nullptr,
       g0p, 64, DM, (long)DM * CSTRIDE, KSPLIT);
  conv_reduce_kernel<<<64, 256, 0, stream>>>(g0p, conv_b, g0);

  // --- global-token MLP ---
  GEMM(float, 2, false, false, g0, DM, gpw116, gp_b1, nullptr, g1, 64, DM, DM, 1);
  GEMM(float, 0, false, false, g1, DM, gpw216, gp_b2, nullptr, gout, 64, DM, DM, 1);
}
